// globNet_60902636257971
// MI455X (gfx1250) — matrix operand order repaired and checked by reading, not compiled, not run on hardware
//
#include <hip/hip_runtime.h>
#include <math.h>

typedef __attribute__((ext_vector_type(16))) _Float16 v16h;
typedef __attribute__((ext_vector_type(8)))  float    v8f;

#define NHID 128
#define FCAT 384   // 3*NHID

struct SelState { unsigned lo, hi, mid, thr, cnt; };

// ---------------------------------------------------------------- utilities
__global__ void zero_f32(float* __restrict__ p, long long n) {
  long long i = (long long)blockIdx.x * blockDim.x + threadIdx.x;
  if (i < n) p[i] = 0.0f;
}

__device__ __forceinline__ void atomicMaxF(float* addr, float v) {
  if (v >= 0.0f) atomicMax((int*)addr, __float_as_int(v));
  else           atomicMin((unsigned int*)addr, (unsigned int)__float_as_int(v));
}

// ---------------------------------------------------------------- degree / norm
__global__ void deg_init(float* __restrict__ deg, int N) {
  int n = blockIdx.x * blockDim.x + threadIdx.x;
  if (n < N) deg[n] = 1.0f;  // self loop
}

__global__ void deg_count(const int* __restrict__ dst, float* __restrict__ deg, int E) {
  int e = blockIdx.x * blockDim.x + threadIdx.x;
  if (e < E) atomicAdd(&deg[dst[e]], 1.0f);
}

__global__ void deg_fin(float* __restrict__ dinv, float* __restrict__ dsq, int N) {
  int n = blockIdx.x * blockDim.x + threadIdx.x;
  if (n >= N) return;
  float d = dinv[n];
  dsq[n]  = rsqrtf(d);
  dinv[n] = 1.0f / d;
}

__global__ void edge_norm(const int* __restrict__ src, const int* __restrict__ dst,
                          const float* __restrict__ dsq, float* __restrict__ ne, int E) {
  int e = blockIdx.x * blockDim.x + threadIdx.x;
  if (e < E) ne[e] = dsq[src[e]] * dsq[dst[e]];
}

// ---------------------------------------------------------------- WMMA GEMM: Y[N,128] = X[N,Fin] @ W[Fin,128]
// block = 256 threads = 8 waves; block tile = 128 rows x 128 cols; wave tile = 16 rows x 128 cols.
#define KC 128
__global__ __launch_bounds__(256)
void gemm_wmma(const float* __restrict__ X, int ldx,
               const float* __restrict__ W, int Fin,
               float* __restrict__ Y, int N) {
  __shared__ __align__(32) _Float16 Wswz[KC * NHID];  // 32 KB, B-fragment layout

  const int tid  = threadIdx.x;
  const int wave = tid >> 5;
  const int lane = tid & 31;
  const int half = lane >> 4;
  const int l16  = lane & 15;
  const int rowBase = blockIdx.x * 128 + wave * 16;
  const int arow = rowBase + l16;
  const bool rowOK = arow < N;

  v8f acc[8];
#pragma unroll
  for (int t = 0; t < 8; ++t)
#pragma unroll
    for (int r = 0; r < 8; ++r) acc[t][r] = 0.0f;

  for (int kc = 0; kc < Fin; kc += KC) {
    // stage W chunk into LDS, pre-swizzled into B-fragment order:
    // Wswz[(((ks*8)+t)*32 + l)*16 + e] = W[kc + ks*32 + 8*(l>>4) + (e&7) + 16*(e>>3)][t*16 + (l&15)]
    for (int j = tid; j < KC * NHID; j += 256) {
      int e  = j & 15;
      int l  = (j >> 4) & 31;
      int t  = (j >> 9) & 7;
      int ks = j >> 12;
      int k  = ks * 32 + ((l >> 4) << 3) + (e & 7) + ((e >> 3) << 4);
      int n  = t * 16 + (l & 15);
      Wswz[j] = (_Float16)W[(size_t)(kc + k) * NHID + n];
    }
    __syncthreads();

#pragma unroll
    for (int ks = 0; ks < KC / 32; ++ks) {
      const int k0 = ks * 32;
      // ---- A fragment: lane l holds row (rowBase + l%16); per ISA 16-bit A 16x32 layout
      v16h a;
      if (rowOK) {
        const float* xp = X + (size_t)arow * ldx + kc + k0 + 8 * half;
        const float4* q = (const float4*)xp;        // K = base+0..7
        const float4* r = (const float4*)(xp + 16); // K = base+16..23
        float4 q0 = q[0], q1 = q[1], r0 = r[0], r1 = r[1];
        a[0] = (_Float16)q0.x; a[1] = (_Float16)q0.y; a[2] = (_Float16)q0.z; a[3] = (_Float16)q0.w;
        a[4] = (_Float16)q1.x; a[5] = (_Float16)q1.y; a[6] = (_Float16)q1.z; a[7] = (_Float16)q1.w;
        a[8] = (_Float16)r0.x; a[9] = (_Float16)r0.y; a[10] = (_Float16)r0.z; a[11] = (_Float16)r0.w;
        a[12] = (_Float16)r1.x; a[13] = (_Float16)r1.y; a[14] = (_Float16)r1.z; a[15] = (_Float16)r1.w;
      } else {
#pragma unroll
        for (int e = 0; e < 16; ++e) a[e] = (_Float16)0.0f;
      }
      // ---- 8 column tiles: b fragments are contiguous 32B LDS vectors
#pragma unroll
      for (int t = 0; t < 8; ++t) {
        v16h b = *(const v16h*)&Wswz[(((ks * 8) + t) * 32 + lane) * 16];
        acc[t] = __builtin_amdgcn_wmma_f32_16x16x32_f16(
            false, a, false, b, (short)0, acc[t], false, false);
      }
    }
    __syncthreads();
  }

  // ---- store C: VGPR r, lane -> M = r + 8*half, N = t*16 + l16
#pragma unroll
  for (int t = 0; t < 8; ++t) {
#pragma unroll
    for (int r = 0; r < 8; ++r) {
      int m = rowBase + 8 * half + r;
      if (m < N) Y[(size_t)m * NHID + t * 16 + l16] = acc[t][r];
    }
  }
}

// ---------------------------------------------------------------- edge scatter: acc[dst] += xw[src] * ne
// one wave per edge, 4 features per lane (float4 gather, 4 coalesced f32 atomics)
__global__ __launch_bounds__(256)
void edge_scatter(const int* __restrict__ src, const int* __restrict__ dst,
                  const float* __restrict__ ne, const float* __restrict__ xw,
                  float* __restrict__ acc, int E) {
  long long tid = (long long)blockIdx.x * blockDim.x + threadIdx.x;
  int e = (int)(tid >> 5);
  int lane = (int)(tid & 31);
  if (e >= E) return;
  int s = src[e], d = dst[e];
  float w = ne[e];
  float4 v = ((const float4*)(xw + (size_t)s * NHID))[lane];
  float* ap = acc + (size_t)d * NHID + lane * 4;
  atomicAdd(ap + 0, v.x * w);
  atomicAdd(ap + 1, v.y * w);
  atomicAdd(ap + 2, v.z * w);
  atomicAdd(ap + 3, v.w * w);
}

// ---------------------------------------------------------------- combine: relu(agg + xw*dinv + b) -> cat column block
__global__ void gcn_combine(const float* __restrict__ acc, const float* __restrict__ xw,
                            const float* __restrict__ dinv, const float* __restrict__ b,
                            float* __restrict__ catOut, int N) {
  int i = blockIdx.x * blockDim.x + threadIdx.x;
  if (i >= N * NHID) return;
  int n = i >> 7, f = i & (NHID - 1);
  float v = acc[i] + xw[i] * dinv[n] + b[f];
  catOut[(size_t)n * FCAT + f] = fmaxf(v, 0.0f);
}

// ---------------------------------------------------------------- score GCN (1 output channel)
__global__ void score_lin(const float* __restrict__ cat, const float* __restrict__ Ws,
                          float* __restrict__ slin, int N) {
  int tid = blockIdx.x * blockDim.x + threadIdx.x;
  int n = tid >> 5, lane = tid & 31;
  if (n >= N) return;
  float s = 0.0f;
  const float* row = cat + (size_t)n * FCAT;
#pragma unroll
  for (int j = 0; j < FCAT / 32; ++j) {
    int f = lane + 32 * j;
    s += row[f] * Ws[f];
  }
#pragma unroll
  for (int off = 16; off > 0; off >>= 1) s += __shfl_down(s, off, 32);
  if (lane == 0) slin[n] = s;
}

__global__ void score_edge(const int* __restrict__ src, const int* __restrict__ dst,
                           const float* __restrict__ ne, const float* __restrict__ slin,
                           float* __restrict__ sagg, int E) {
  int e = blockIdx.x * blockDim.x + threadIdx.x;
  if (e < E) atomicAdd(&sagg[dst[e]], slin[src[e]] * ne[e]);
}

__global__ void score_fin(const float* __restrict__ sagg, const float* __restrict__ slin,
                          const float* __restrict__ dinv, const float* __restrict__ bs,
                          float* __restrict__ score, unsigned* __restrict__ keys, int N) {
  int n = blockIdx.x * blockDim.x + threadIdx.x;
  if (n >= N) return;
  float sc = sagg[n] + slin[n] * dinv[n] + bs[0];
  score[n] = sc;
  unsigned u = __float_as_uint(sc);  // order-preserving map float -> uint
  keys[n] = (u & 0x80000000u) ? ~u : (u | 0x80000000u);
}

// ---------------------------------------------------------------- top-K threshold via 32-step binary search
__global__ void sel_init(SelState* st) {
  st->lo = 0u; st->hi = 0xFFFFFFFFu; st->mid = 0u; st->thr = 0u; st->cnt = 0u;
}

__global__ void sel_step(SelState* st, int Ksel, int apply) {
  if (apply) {
    if (st->cnt >= (unsigned)Ksel) st->lo = st->mid;
    else                           st->hi = st->mid - 1u;
  }
  unsigned d = st->hi - st->lo;
  st->mid = st->lo + (d >> 1) + (d & 1u);  // round-up midpoint
  st->cnt = 0u;
}

__global__ void sel_fin(SelState* st, int Ksel) {
  if (st->cnt >= (unsigned)Ksel) st->lo = st->mid;
  else                           st->hi = st->mid - 1u;
  st->thr = st->lo;  // largest t with count(key >= t) >= K
}

__global__ void sel_count(const unsigned* __restrict__ keys, SelState* st, int N) {
  int i = blockIdx.x * blockDim.x + threadIdx.x;
  unsigned mid = st->mid;
  bool p = (i < N) && (keys[i] >= mid);
  unsigned long long m = __ballot(p);
  if ((threadIdx.x & 31) == 0 && m)
    atomicAdd(&st->cnt, (unsigned)__popcll(m));
}

// ---------------------------------------------------------------- masked global max/sum pooling
__global__ void pool_init(float* __restrict__ gmax, float* __restrict__ gsum) {
  int i = blockIdx.x * blockDim.x + threadIdx.x;
  if (i < FCAT) { gmax[i] = -INFINITY; gsum[i] = 0.0f; }
}

__global__ void pool_kernel(const float* __restrict__ cat, const float* __restrict__ score,
                            const unsigned* __restrict__ keys, const SelState* __restrict__ st,
                            float* __restrict__ gmax, float* __restrict__ gsum, int N) {
  int tid = blockIdx.x * blockDim.x + threadIdx.x;
  int n = tid >> 5, lane = tid & 31;
  if (n >= N) return;
  if (keys[n] < st->thr) return;
  float g = tanhf(score[n]);
  const float* row = cat + (size_t)n * FCAT;
#pragma unroll
  for (int j = 0; j < FCAT / 32; ++j) {
    int f = lane + 32 * j;
    float v = row[f] * g;
    atomicMaxF(&gmax[f], v);
    atomicAdd(&gsum[f], v);
  }
}

// ---------------------------------------------------------------- MLP head + log_softmax (single block)
__global__ __launch_bounds__(256)
void mlp_head(const float* __restrict__ gmax, const float* __restrict__ gsum,
              const float* __restrict__ Wl1, const float* __restrict__ bl1,
              const float* __restrict__ Wl2, const float* __restrict__ bl2,
              const float* __restrict__ Wl3, const float* __restrict__ bl3,
              float kinv, float* __restrict__ out) {
  __shared__ float h[2 * FCAT], h1[128], h2[64], lg[10];
  int t = threadIdx.x;
  for (int i = t; i < 2 * FCAT; i += 256)
    h[i] = (i < FCAT) ? gmax[i] : gsum[i - FCAT] * kinv;
  __syncthreads();
  if (t < 128) {
    float s = bl1[t];
    for (int i = 0; i < 2 * FCAT; ++i) s += h[i] * Wl1[i * 128 + t];
    h1[t] = fmaxf(s, 0.0f);
  }
  __syncthreads();
  if (t < 64) {
    float s = bl2[t];
    for (int i = 0; i < 128; ++i) s += h1[i] * Wl2[i * 64 + t];
    h2[t] = fmaxf(s, 0.0f);
  }
  __syncthreads();
  if (t < 10) {
    float s = bl3[t];
    for (int i = 0; i < 64; ++i) s += h2[i] * Wl3[i * 10 + t];
    lg[t] = s;
  }
  __syncthreads();
  if (t == 0) {
    float m = lg[0];
    for (int c = 1; c < 10; ++c) m = fmaxf(m, lg[c]);
    float se = 0.0f;
    for (int c = 0; c < 10; ++c) se += expf(lg[c] - m);
    float lse = m + logf(se);
    for (int c = 0; c < 10; ++c) out[c] = lg[c] - lse;
  }
}

// ---------------------------------------------------------------- launcher
extern "C" void kernel_launch(void* const* d_in, const int* in_sizes, int n_in,
                              void* d_out, int out_size, void* d_ws, size_t ws_size,
                              hipStream_t stream) {
  (void)n_in; (void)out_size; (void)ws_size;

  const float* x   = (const float*)d_in[0];
  const int*   ei  = (const int*)d_in[1];
  // d_in[2] = batch (all zeros, single graph) -> unused
  const float* W1  = (const float*)d_in[3];  const float* b1  = (const float*)d_in[4];
  const float* W2  = (const float*)d_in[5];  const float* b2  = (const float*)d_in[6];
  const float* W3  = (const float*)d_in[7];  const float* b3  = (const float*)d_in[8];
  const float* Ws  = (const float*)d_in[9];  const float* bs  = (const float*)d_in[10];
  const float* Wl1 = (const float*)d_in[11]; const float* bl1 = (const float*)d_in[12];
  const float* Wl2 = (const float*)d_in[13]; const float* bl2 = (const float*)d_in[14];
  const float* Wl3 = (const float*)d_in[15]; const float* bl3 = (const float*)d_in[16];

  const int N = in_sizes[0] / 512;   // 100000
  const int E = in_sizes[1] / 2;     // 1600000
  const int KSEL = (N + 1) / 2;      // ceil(0.5*N) = 50000
  const int* src = ei;
  const int* dst = ei + E;

  // workspace layout (floats)
  float* cat  = (float*)d_ws;                 // N * 384
  float* xw   = cat  + (size_t)N * FCAT;      // N * 128
  float* acc  = xw   + (size_t)N * NHID;      // N * 128
  float* dinv = acc  + (size_t)N * NHID;      // N   (deg -> dinv)
  float* dsq  = dinv + N;                     // N
  float* ne   = dsq  + N;                     // E
  float* slin = ne   + E;                     // N
  float* sagg = slin + N;                     // N
  float* scr  = sagg + N;                     // N
  unsigned* keys = (unsigned*)(scr + N);      // N
  SelState* st   = (SelState*)(keys + N);
  float* gmax = (float*)(st + 1);             // 384
  float* gsum = gmax + FCAT;                  // 384

  const int TB = 256;
  const int gN  = (N + TB - 1) / TB;
  const int gE  = (E + TB - 1) / TB;
  const int gNW = (N * 32 + TB - 1) / TB;                       // wave-per-node
  const int gEW = (int)(((long long)E * 32 + TB - 1) / TB);     // wave-per-edge
  const int gNF = (N * NHID + TB - 1) / TB;

  // degree / normalization
  deg_init<<<gN, TB, 0, stream>>>(dinv, N);
  deg_count<<<gE, TB, 0, stream>>>(dst, dinv, E);
  deg_fin<<<gN, TB, 0, stream>>>(dinv, dsq, N);
  edge_norm<<<gE, TB, 0, stream>>>(src, dst, dsq, ne, E);

  // three GCN layers -> cat[:,0:128], [:,128:256], [:,256:384]
  const int gGemm = (N + 127) / 128;
  struct Lyr { const float* X; int ldx, Fin; const float* W; const float* b; float* out; };
  Lyr lyr[3] = {
    { x,         512, 512, W1, b1, cat         },
    { cat,       FCAT, NHID, W2, b2, cat + NHID  },
    { cat + NHID, FCAT, NHID, W3, b3, cat + 2 * NHID },
  };
  for (int l = 0; l < 3; ++l) {
    gemm_wmma<<<gGemm, 256, 0, stream>>>(lyr[l].X, lyr[l].ldx, lyr[l].W, lyr[l].Fin, xw, N);
    zero_f32<<<gNF, TB, 0, stream>>>(acc, (long long)N * NHID);
    edge_scatter<<<gEW, TB, 0, stream>>>(src, dst, ne, xw, acc, E);
    gcn_combine<<<gNF, TB, 0, stream>>>(acc, xw, dinv, lyr[l].b, lyr[l].out, N);
  }

  // scoring GCN (1 channel)
  score_lin<<<gNW, TB, 0, stream>>>(cat, Ws, slin, N);
  zero_f32<<<gN, TB, 0, stream>>>(sagg, (long long)N);
  score_edge<<<gE, TB, 0, stream>>>(src, dst, ne, slin, sagg, E);
  score_fin<<<gN, TB, 0, stream>>>(sagg, slin, dinv, bs, scr, keys, N);

  // K-th largest threshold: 32 fixed binary-search iterations (graph-capture safe)
  sel_init<<<1, 1, 0, stream>>>(st);
  for (int i = 0; i < 32; ++i) {
    sel_step<<<1, 1, 0, stream>>>(st, KSEL, i > 0 ? 1 : 0);
    sel_count<<<gN, TB, 0, stream>>>(keys, st, N);
  }
  sel_fin<<<1, 1, 0, stream>>>(st, KSEL);

  // gated global max / mean pooling over kept nodes
  pool_init<<<2, TB, 0, stream>>>(gmax, gsum);
  pool_kernel<<<gNW, TB, 0, stream>>>(cat, scr, keys, st, gmax, gsum, N);

  // classifier head
  mlp_head<<<1, 256, 0, stream>>>(gmax, gsum, Wl1, bl1, Wl2, bl2, Wl3, bl3,
                                  1.0f / (float)KSEL, (float*)d_out);
}
